// FeedBack_15247133900820
// MI455X (gfx1250) — matrix operand order repaired and checked by reading, not compiled, not run on hardware
//
#include <hip/hip_runtime.h>

typedef _Float16 f16_t;
typedef __attribute__((ext_vector_type(8)))  _Float16 v8h;
typedef __attribute__((ext_vector_type(16))) _Float16 v16h;
typedef __attribute__((ext_vector_type(8)))  float    v8f;

#define B_TOTAL   4096
#define T_IN      128
#define FEAT      12
#define UNITS     512
#define NGATE     2048
#define OUT_STEPS 32
#define RROWS     64      // batch rows per workgroup
#define TPB       512     // 16 waves (wave32)

union Frag32 { v16h v; v8h h[2]; };

// B-fragment (32x16 f16, K x N): lane n=lane%16 holds K {0..7,16..23} (lanes 0-15)
// or K {8..15,24..31} (lanes 16-31). p = row base + kc + lhalf*8.
__device__ __forceinline__ v16h load_b32(const f16_t* p) {
  Frag32 f;
  f.h[0] = *(const v8h*)(p);
  f.h[1] = *(const v8h*)(p + 16);
  return f.v;
}

// A-fragment (16x32 f16, M x K): lane<16 row M=lane holds K {0..7,16..23},
// lane>=16 row M=lane-16 holds K {8..15,24..31}. p = row base + kc + lhalf*8.
__device__ __forceinline__ v16h load_a32(const f16_t* p) {
  Frag32 f;
  f.h[0] = *(const v8h*)(p);
  f.h[1] = *(const v8h*)(p + 16);
  return f.v;
}

__device__ __forceinline__ float fsig(float x)  { return 1.0f / (1.0f + __expf(-x)); }
__device__ __forceinline__ float ftanh(float x) { return 2.0f / (1.0f + __expf(-2.0f * x)) - 1.0f; }

// pred = h @ Wd + bd for one 16-row M-tile; writes fp32 to out[:, d, :] and
// f16 into the (zero-padded, 32-wide) x staging buffer for feedback.
__device__ __forceinline__ void compute_pred(
    const f16_t* __restrict__ hc, const f16_t* __restrict__ Wdt,
    const float* __restrict__ bd, float* __restrict__ out,
    f16_t* __restrict__ xdst, int row0, int d, int m, int lhalf, int l16)
{
  v8f acc;
  float bb = (l16 < FEAT) ? bd[l16] : 0.0f;
#pragma unroll
  for (int j = 0; j < 8; ++j) acc[j] = bb;
#pragma unroll 4
  for (int kc = 0; kc < UNITS; kc += 32) {
    v16h af  = load_a32(hc + (m * 16 + l16) * UNITS + kc + lhalf * 8);
    v16h bfg = load_b32(Wdt + l16 * UNITS + kc + lhalf * 8);
    acc = __builtin_amdgcn_wmma_f32_16x16x32_f16(false, af, false, bfg,
                                                 (short)0, acc, false, false);
  }
#pragma unroll
  for (int j = 0; j < 8; ++j) {
    int r = m * 16 + lhalf * 8 + j;  // C layout: VGPR j -> M = j + 8*lhalf
    if (l16 < FEAT) {
      out[((size_t)(row0 + r) * OUT_STEPS + d) * FEAT + l16] = acc[j];
      xdst[r * 32 + l16] = (f16_t)acc[j];
    }
  }
}

__global__ __launch_bounds__(TPB, 1) void lstm_feedback_kernel(
    const float* __restrict__ x,    // [4096,128,12]
    const float* __restrict__ b,    // [2048]
    const float* __restrict__ bd,   // [12]
    const f16_t* __restrict__ Ut,   // [2048][512]  (U^T, n-major)
    const f16_t* __restrict__ Wt,   // [2048][32]   (W^T, K zero-padded 12->32)
    const f16_t* __restrict__ Wdt,  // [16][512]    (Wd^T, N zero-padded 12->16)
    float* __restrict__ out)        // [4096][32][12]
{
  extern __shared__ f16_t smem[];
  f16_t* hbuf = smem;                          // 2 * RROWS * UNITS (double buf)
  f16_t* xbuf = smem + 2 * RROWS * UNITS;      // 2 * RROWS * 32    (double buf)

  const int tid    = threadIdx.x;
  const int wave   = tid >> 5;
  const int lane   = tid & 31;
  const int l16    = lane & 15;
  const int lhalf  = lane >> 4;
  const int mtile  = wave & 3;          // 4 M-groups of 16 rows
  const int utbase = (wave >> 2) * 8;   // 8 unit-tiles (16 units each) per wave
  const int row0   = blockIdx.x * RROWS;

  // zero h (h0 = 0) and x staging (incl. the zero padding cols 12..31)
  for (int i = tid; i < 2 * RROWS * UNITS; i += TPB) hbuf[i] = (f16_t)0.0f;
  for (int i = tid; i < 2 * RROWS * 32;   i += TPB) xbuf[i] = (f16_t)0.0f;
  __syncthreads();
  for (int i = tid; i < RROWS * FEAT; i += TPB) {
    int r = i / FEAT, cc = i % FEAT;
    xbuf[r * 32 + cc] = (f16_t)x[((size_t)(row0 + r) * T_IN + 0) * FEAT + cc];
  }
  __syncthreads();

  // step-invariant: bias per (unit-tile, gate) for this lane's column
  float bias[8][4];
#pragma unroll
  for (int u = 0; u < 8; ++u)
#pragma unroll
    for (int g = 0; g < 4; ++g)
      bias[u][g] = b[g * UNITS + (utbase + u) * 16 + l16];

  // c state lives in VGPRs: 8 unit-tiles x v8f
  v8f cst[8];
#pragma unroll
  for (int u = 0; u < 8; ++u)
#pragma unroll
    for (int j = 0; j < 8; ++j) cst[u][j] = 0.0f;

  const int TOTAL = T_IN + OUT_STEPS - 1;  // 159 LSTM cell evaluations
#pragma unroll 1
  for (int s = 0; s < TOTAL; ++s) {
    const int cur = s & 1;
    const f16_t* hc = hbuf + cur * RROWS * UNITS;
    f16_t*       hn = hbuf + (cur ^ 1) * RROWS * UNITS;
    f16_t*       xc = xbuf + cur * RROWS * 32;
    f16_t*       xn = xbuf + (cur ^ 1) * RROWS * 32;

    // decode: output pred_{s-128} from current h and feed it back as x_s
    if (s >= T_IN) {
      if (wave < 4)
        compute_pred(hc, Wdt, bd, out, xc, row0, s - T_IN, wave, lhalf, l16);
      __syncthreads();
    }

    // x A-fragment for this wave's row tile (32-wide K, zero padded past 12)
    v16h xfrag = load_a32(xc + (mtile * 16 + l16) * 32 + lhalf * 8);

#pragma unroll 1
    for (int u = 0; u < 8; ++u) {
      const int ut   = utbase + u;
      const int ncol = ut * 16 + l16;  // this lane's unit column

      // acc init: bias + x @ W  (one K=32 WMMA per gate)
      v8f acc[4];
#pragma unroll
      for (int g = 0; g < 4; ++g) {
        float bb = bias[u][g];
        v8f ci;
#pragma unroll
        for (int j = 0; j < 8; ++j) ci[j] = bb;
        v16h wfrag = load_b32(Wt + (size_t)(g * UNITS + ncol) * 32 + lhalf * 8);
        acc[g] = __builtin_amdgcn_wmma_f32_16x16x32_f16(false, xfrag, false, wfrag,
                                                        (short)0, ci, false, false);
      }

      // z += h @ U : A-frag shared across the 4 gates, B streamed from L2
      const f16_t* arow = hc + (mtile * 16 + l16) * UNITS;
#pragma unroll 2
      for (int kc = 0; kc < UNITS; kc += 32) {
        v16h af = load_a32(arow + kc + lhalf * 8);
#pragma unroll
        for (int g = 0; g < 4; ++g) {
          v16h ufrag = load_b32(Ut + (size_t)(g * UNITS + ncol) * UNITS + kc + lhalf * 8);
          acc[g] = __builtin_amdgcn_wmma_f32_16x16x32_f16(false, af, false, ufrag,
                                                          (short)0, acc[g], false, false);
        }
      }

      // fused gates (Keras order i,f,g,o) + state update, h -> LDS (f16)
      v8f cv = cst[u];
#pragma unroll
      for (int j = 0; j < 8; ++j) {
        float iv = fsig(acc[0][j]);
        float fv = fsig(acc[1][j]);
        float gv = ftanh(acc[2][j]);
        float ov = fsig(acc[3][j]);
        float cn = fv * cv[j] + iv * gv;
        cv[j] = cn;
        float hv = ov * ftanh(cn);
        hn[(mtile * 16 + lhalf * 8 + j) * UNITS + ncol] = (f16_t)hv;
      }
      cst[u] = cv;
    }

    // warmup: stage next timestep's x while the GEMM results drain
    if (s + 1 < T_IN) {
      for (int i = tid; i < RROWS * FEAT; i += TPB) {
        int r = i / FEAT, cc = i % FEAT;
        xn[r * 32 + cc] = (f16_t)x[((size_t)(row0 + r) * T_IN + (s + 1)) * FEAT + cc];
      }
    }
    __syncthreads();
  }

  // final prediction (d = 31) from the final hidden state
  {
    const f16_t* hfin = hbuf + (TOTAL & 1) * RROWS * UNITS;
    if (wave < 4)
      compute_pred(hfin, Wdt, bd, out, xbuf, row0, OUT_STEPS - 1, wave, lhalf, l16);
  }
}

// ---- prolog: f16 weight transposes/padding into workspace ----

__global__ void prep_U_kernel(const float* __restrict__ U, f16_t* __restrict__ Ut) {
  int idx = blockIdx.x * 256 + threadIdx.x;           // Ut[n][k] = U[k][n]
  if (idx < NGATE * UNITS) {
    int n = idx / UNITS, k = idx % UNITS;
    Ut[idx] = (f16_t)U[(size_t)k * NGATE + n];
  }
}
__global__ void prep_W_kernel(const float* __restrict__ W, f16_t* __restrict__ Wt) {
  int idx = blockIdx.x * 256 + threadIdx.x;           // Wt[n][kk], kk<12 real
  if (idx < NGATE * 32) {
    int n = idx / 32, k = idx % 32;
    Wt[idx] = (k < FEAT) ? (f16_t)W[(size_t)k * NGATE + n] : (f16_t)0.0f;
  }
}
__global__ void prep_Wd_kernel(const float* __restrict__ Wd, f16_t* __restrict__ Wdt) {
  int idx = blockIdx.x * 256 + threadIdx.x;           // Wdt[n][k], n<12 real
  if (idx < 16 * UNITS) {
    int n = idx / UNITS, k = idx % UNITS;
    Wdt[idx] = (n < FEAT) ? (f16_t)Wd[(size_t)k * FEAT + n] : (f16_t)0.0f;
  }
}

extern "C" void kernel_launch(void* const* d_in, const int* in_sizes, int n_in,
                              void* d_out, int out_size, void* d_ws, size_t ws_size,
                              hipStream_t stream) {
  (void)in_sizes; (void)n_in; (void)out_size; (void)ws_size;
  const float* x  = (const float*)d_in[0];
  const float* W  = (const float*)d_in[1];
  const float* U  = (const float*)d_in[2];
  const float* b  = (const float*)d_in[3];
  const float* Wd = (const float*)d_in[4];
  const float* bd = (const float*)d_in[5];
  float* out = (float*)d_out;

  f16_t* Ut  = (f16_t*)d_ws;              // 2048*512 f16 = 2 MB
  f16_t* Wt  = Ut + NGATE * UNITS;        // 2048*32  f16 = 128 KB
  f16_t* Wdt = Wt + NGATE * 32;           // 16*512   f16 = 16 KB

  prep_U_kernel <<<(NGATE * UNITS + 255) / 256, 256, 0, stream>>>(U, Ut);
  prep_W_kernel <<<(NGATE * 32 + 255) / 256,   256, 0, stream>>>(W, Wt);
  prep_Wd_kernel<<<(16 * UNITS + 255) / 256,   256, 0, stream>>>(Wd, Wdt);

  size_t shmem = (size_t)(2 * RROWS * UNITS + 2 * RROWS * 32) * sizeof(f16_t); // 136 KB
  lstm_feedback_kernel<<<B_TOTAL / RROWS, TPB, shmem, stream>>>(x, b, bd, Ut, Wt, Wdt, out);
}
